// MHA_44684839748002
// MI455X (gfx1250) — matrix operand order repaired and checked by reading, not compiled, not run on hardware
//
#include <hip/hip_runtime.h>
#include <hip/hip_bf16.h>

// ---------------------------------------------------------------------------
// MI455X (gfx1250) wave32 WMMA implementation of the RMSNorm+MHA+residual op.
// bf16 inputs to v_wmma_f32_16x16x32_bf16, f32 accumulate. GEMMs are fed from
// global (entire working set is resident in the 192MB L2) with a double-
// buffered software pipeline so WMMAs overlap the fragment loads.
// ---------------------------------------------------------------------------

typedef __attribute__((ext_vector_type(16))) __bf16          v16bf;
typedef __attribute__((ext_vector_type(8)))  float           v8f;
typedef __attribute__((ext_vector_type(8)))  unsigned short  u16x8;

union BF16x16 {
    struct { u16x8 lo, hi; } h;
    v16bf v;
};

__device__ __forceinline__ unsigned short f2bfu(float f) {
    unsigned u = __float_as_uint(f);
    unsigned r = u + 0x7FFFu + ((u >> 16) & 1u);   // round-to-nearest-even
    return (unsigned short)(r >> 16);
}

__device__ __forceinline__ v8f wmma_bf16(const BF16x16& a, const BF16x16& b, v8f c) {
    // 8 args: (neg_a, A, neg_b, B, c_mod, C, reuse_a, reuse_b)
    return __builtin_amdgcn_wmma_f32_16x16x32_bf16(
        false, a.v, false, b.v, (short)0, c, false, false);
}

#define D_MODEL 2048
#define NHEAD   16
#define DK      128

// ---------------------------------------------------------------------------
// 1) f32 -> bf16 weight cast
// ---------------------------------------------------------------------------
__global__ void __launch_bounds__(256)
cast_f32_bf16(const float* __restrict__ in, unsigned short* __restrict__ out, int n)
{
    int i = blockIdx.x * 256 + threadIdx.x;
    if (i < n) out[i] = f2bfu(in[i]);
}

// ---------------------------------------------------------------------------
// 2) RMSNorm (per row of D_MODEL) -> bf16
// ---------------------------------------------------------------------------
__global__ void __launch_bounds__(256)
rmsnorm_to_bf16(const float* __restrict__ X, const float* __restrict__ w,
                unsigned short* __restrict__ Xn, int D)
{
    const int row = blockIdx.x;
    const float* xr = X + (size_t)row * D;
    float ss = 0.f;
    for (int d = threadIdx.x; d < D; d += 256) { float v = xr[d]; ss += v * v; }
#pragma unroll
    for (int m = 16; m >= 1; m >>= 1) ss += __shfl_xor(ss, m, 32);
    __shared__ float wsum[8];
    if ((threadIdx.x & 31) == 0) wsum[threadIdx.x >> 5] = ss;
    __syncthreads();
    float tot = 0.f;
#pragma unroll
    for (int i = 0; i < 8; ++i) tot += wsum[i];
    const float scale = rsqrtf(tot / (float)D + 1.1920929e-07f);
    unsigned short* xo = Xn + (size_t)row * D;
    for (int d = threadIdx.x; d < D; d += 256)
        xo[d] = f2bfu(xr[d] * scale * w[d]);
}

// ---------------------------------------------------------------------------
// 3/6) C[M,N] = A[M,K](bf16) * W[N,K]^T(bf16) + bias (+ residual), f32 out.
//      8 waves/block arranged 2Mx4N; each wave owns a 32Mx64N stripe:
//      2 A-fragments x 4 B-fragments -> 8 accumulators per k-step.
//      Double-buffered (ping-pong fragment sets, K unrolled x2) so the
//      v_wmma stream overlaps the global_load_b128 stream.
// ---------------------------------------------------------------------------
__global__ void __launch_bounds__(256)
gemm_bf16_nt(const unsigned short* __restrict__ A,   // M x K
             const unsigned short* __restrict__ W,   // N x K
             const float* __restrict__ bias,         // N
             const float* __restrict__ resid,        // M x N (may be null)
             float* __restrict__ C,                  // M x N
             int M, int N, int K, int add_resid)
{
    const int lane  = threadIdx.x & 31;
    const int wave  = threadIdx.x >> 5;
    const int waveM = wave >> 2;                 // 0..1
    const int waveN = wave & 3;                  // 0..3
    const int m0 = blockIdx.y * 64 + waveM * 32;
    const int n0 = blockIdx.x * 256 + waveN * 64;

    const int lm = lane & 15;
    const int lh = lane >> 4;

    // per-lane row pointers (A rows for the two 16-row subtiles; W rows for
    // the four 16-col subtiles; B-fragment per-lane data is 32B contiguous)
    const unsigned short* Arow0 = A + (size_t)(m0 + lm) * K;
    const unsigned short* Arow1 = A + (size_t)(m0 + 16 + lm) * K;
    const unsigned short* Wrow0 = W + (size_t)(n0 + 0 * 16 + lm) * K + lh * 16;
    const unsigned short* Wrow1 = W + (size_t)(n0 + 1 * 16 + lm) * K + lh * 16;
    const unsigned short* Wrow2 = W + (size_t)(n0 + 2 * 16 + lm) * K + lh * 16;
    const unsigned short* Wrow3 = W + (size_t)(n0 + 3 * 16 + lm) * K + lh * 16;

    v8f acc[2][4];
#pragma unroll
    for (int r = 0; r < 2; ++r)
#pragma unroll
        for (int t = 0; t < 4; ++t) acc[r][t] = v8f{};

    BF16x16 a0[2], b0[4], a1[2], b1[4];

    auto load_set = [&](BF16x16 (&a)[2], BF16x16 (&b)[4], int k0) {
        a[0].h.lo = *(const u16x8*)(Arow0 + k0 + lh * 16);
        a[0].h.hi = *(const u16x8*)(Arow0 + k0 + lh * 16 + 8);
        a[1].h.lo = *(const u16x8*)(Arow1 + k0 + lh * 16);
        a[1].h.hi = *(const u16x8*)(Arow1 + k0 + lh * 16 + 8);
        b[0].h.lo = *(const u16x8*)(Wrow0 + k0);
        b[0].h.hi = *(const u16x8*)(Wrow0 + k0 + 8);
        b[1].h.lo = *(const u16x8*)(Wrow1 + k0);
        b[1].h.hi = *(const u16x8*)(Wrow1 + k0 + 8);
        b[2].h.lo = *(const u16x8*)(Wrow2 + k0);
        b[2].h.hi = *(const u16x8*)(Wrow2 + k0 + 8);
        b[3].h.lo = *(const u16x8*)(Wrow3 + k0);
        b[3].h.hi = *(const u16x8*)(Wrow3 + k0 + 8);
    };
    auto mma_set = [&](const BF16x16 (&a)[2], const BF16x16 (&b)[4]) {
#pragma unroll
        for (int r = 0; r < 2; ++r)
#pragma unroll
            for (int t = 0; t < 4; ++t)
                acc[r][t] = wmma_bf16(a[r], b[t], acc[r][t]);
    };

    // software pipeline: K is a multiple of 64
    load_set(a0, b0, 0);
    for (int k0 = 0; k0 < K; k0 += 64) {
        if (k0 + 128 < K) {                      // global_prefetch_b8 ahead
            __builtin_prefetch(Arow0 + k0 + 128, 0, 1);
            __builtin_prefetch(Wrow0 + k0 + 128, 0, 1);
        }
        load_set(a1, b1, k0 + 32);               // fill buffer 1
        mma_set(a0, b0);                         // consume buffer 0
        if (k0 + 64 < K) load_set(a0, b0, k0 + 64);
        mma_set(a1, b1);                         // consume buffer 1
    }

    const int mrow = (lane < 16) ? 0 : 8;
#pragma unroll
    for (int r = 0; r < 2; ++r) {
#pragma unroll
        for (int t = 0; t < 4; ++t) {
            const int n = n0 + t * 16 + lm;
            const float bn = bias[n];
#pragma unroll
            for (int i = 0; i < 8; ++i) {
                const int m = m0 + r * 16 + mrow + i;
                float v = acc[r][t][i] + bn;
                if (add_resid) v += resid[(size_t)m * N + n];
                C[(size_t)m * N + n] = v;
            }
        }
    }
}

// ---------------------------------------------------------------------------
// 4) RoPE + repack: Q f32 [B*S, 2048] -> Qr,V bf16 [B*H, S, 128]
// ---------------------------------------------------------------------------
__global__ void __launch_bounds__(256)
rope_pack(const float* __restrict__ Q, unsigned short* __restrict__ Qr,
          unsigned short* __restrict__ Vb, int S)
{
    const size_t idx = (size_t)blockIdx.x * 256 + threadIdx.x;   // B*S*2048
    const int col = (int)(idx & (D_MODEL - 1));
    const size_t row = idx >> 11;                                // b*S + s
    const int s = (int)(row % (size_t)S);
    const int b = (int)(row / (size_t)S);
    const int h = col >> 7;
    const int d = col & (DK - 1);
    const int fi = d & 63;

    const float q   = Q[idx];
    const float rot = (d < 64) ? -Q[idx + 64] : Q[idx - 64];
    // inv_freq = 10000^(-fi/64) = exp(-fi * ln(10000)/64)
    const float ang = (float)s * __expf(-0.14391157f * (float)fi);
    float sn, cs;
    __sincosf(ang, &sn, &cs);

    const size_t o = (((size_t)(b * NHEAD + h)) * S + s) * DK + d;
    Qr[o] = f2bfu(q * cs + rot * sn);
    Vb[o] = f2bfu(q);
}

// ---------------------------------------------------------------------------
// 5) Flash attention (causal). One wave32 per (b, h, 16-query tile).
//    Kr == Qr, V is pre-RoPE Q (per the reference's Q=K=V behavior).
//    Output written as bf16 in [B*S, 2048] layout for the Wo GEMM.
// ---------------------------------------------------------------------------
__global__ void __launch_bounds__(32)
flash_attn(const unsigned short* __restrict__ Qr,   // [B*H, S, 128]
           const unsigned short* __restrict__ Vb,   // [B*H, S, 128]
           unsigned short* __restrict__ O,          // [B*S, 2048]
           int S)
{
    __shared__ __align__(16) unsigned short Plds[16][32];   // 16x32 P tile
    __shared__ __align__(16) unsigned short Vt[128][32];    // V^T: [d][k]

    const int lane  = threadIdx.x;
    const int bh    = blockIdx.y;
    const int b     = bh >> 4;
    const int h     = bh & 15;
    const int qbase = blockIdx.x * 16;

    const int lm = lane & 15;
    const int lh = lane >> 4;

    const unsigned short* Qh = Qr + (size_t)bh * S * DK;
    const unsigned short* Vh = Vb + (size_t)bh * S * DK;

    // Q held in registers as 4 A-fragments (Dk = 4 * 32)
    BF16x16 qf[4];
#pragma unroll
    for (int c = 0; c < 4; ++c) {
        const unsigned short* p = Qh + (size_t)(qbase + lm) * DK + c * 32;
        qf[c].h.lo = *(const u16x8*)(p + lh * 16);
        qf[c].h.hi = *(const u16x8*)(p + lh * 16 + 8);
    }

    float mi[8], li[8];
    v8f o[8];
#pragma unroll
    for (int i = 0; i < 8; ++i) { mi[i] = -3.0e38f; li[i] = 0.f; }
#pragma unroll
    for (int t = 0; t < 8; ++t) o[t] = v8f{};

    const float sc = 0.08838834764831845f;   // 1/sqrt(128)

    for (int jb = 0; jb <= qbase + 15; jb += 32) {
        // prefetch next key block's K and V rows
        if (jb + 32 <= qbase + 15) {
            __builtin_prefetch(Qh + (size_t)(jb + 32 + lane) * DK, 0, 1);
            __builtin_prefetch(Vh + (size_t)(jb + 32 + lane) * DK, 0, 1);
        }

        // ---- scores: issue all 8 K fragments, then 8 WMMAs ----
        BF16x16 kf[8];
#pragma unroll
        for (int c = 0; c < 4; ++c) {
            const unsigned short* p0 = Qh + (size_t)(jb + lm) * DK + c * 32 + lh * 16;
            const unsigned short* p1 = Qh + (size_t)(jb + 16 + lm) * DK + c * 32 + lh * 16;
            kf[c].h.lo     = *(const u16x8*)(p0);
            kf[c].h.hi     = *(const u16x8*)(p0 + 8);
            kf[4 + c].h.lo = *(const u16x8*)(p1);
            kf[4 + c].h.hi = *(const u16x8*)(p1 + 8);
        }
        v8f s0 = v8f{}, s1 = v8f{};
#pragma unroll
        for (int c = 0; c < 4; ++c) {
            s0 = wmma_bf16(qf[c], kf[c], s0);
            s1 = wmma_bf16(qf[c], kf[4 + c], s1);
        }

        // ---- causal mask + online softmax (rows live in 16-lane groups) ----
        const int kn0 = jb + lm, kn1 = jb + 16 + lm;
        float p0v[8], p1v[8], rmax[8];
#pragma unroll
        for (int i = 0; i < 8; ++i) {
            const int qm = qbase + i + (lh << 3);
            float a0 = (kn0 <= qm) ? s0[i] * sc : -3.0e38f;
            float a1 = (kn1 <= qm) ? s1[i] * sc : -3.0e38f;
            p0v[i] = a0; p1v[i] = a1;
            float r = fmaxf(a0, a1);
            r = fmaxf(r, __shfl_xor(r, 1, 32));
            r = fmaxf(r, __shfl_xor(r, 2, 32));
            r = fmaxf(r, __shfl_xor(r, 4, 32));
            r = fmaxf(r, __shfl_xor(r, 8, 32));
            rmax[i] = r;
        }
#pragma unroll
        for (int i = 0; i < 8; ++i) {
            const float mn = fmaxf(mi[i], rmax[i]);
            const float alpha = __expf(mi[i] - mn);
            const float e0 = __expf(p0v[i] - mn);
            const float e1 = __expf(p1v[i] - mn);
            float rs = e0 + e1;
            rs += __shfl_xor(rs, 1, 32);
            rs += __shfl_xor(rs, 2, 32);
            rs += __shfl_xor(rs, 4, 32);
            rs += __shfl_xor(rs, 8, 32);
            li[i] = li[i] * alpha + rs;
            mi[i] = mn;
#pragma unroll
            for (int t = 0; t < 8; ++t) o[t][i] *= alpha;
            // stage P (C-layout -> LDS, row-major 16x32)
            Plds[i + (lh << 3)][lm]      = f2bfu(e0);
            Plds[i + (lh << 3)][16 + lm] = f2bfu(e1);
        }
        __syncthreads();

        // ---- stage V^T for this 32-key block (coalesced in, transposed out)
        const unsigned short* vrow = Vh + (size_t)(jb + lane) * DK;
#pragma unroll
        for (int c = 0; c < 16; ++c) {
            u16x8 v8 = *(const u16x8*)(vrow + c * 8);
#pragma unroll
            for (int e = 0; e < 8; ++e) Vt[c * 8 + e][lane] = v8[e];
        }
        __syncthreads();

        // ---- O += P(16x32) * V(32x16) per 16-wide d tile ----
        BF16x16 pf;     // re-read P in A-fragment layout
        pf.h.lo = *(const u16x8*)(&Plds[lm][lh * 16]);
        pf.h.hi = *(const u16x8*)(&Plds[lm][lh * 16 + 8]);
#pragma unroll
        for (int t = 0; t < 8; ++t) {
            BF16x16 vf; // B fragment: lane holds Vt[d][16*lh .. +15] contiguous
            vf.h.lo = *(const u16x8*)(&Vt[t * 16 + lm][lh * 16]);
            vf.h.hi = *(const u16x8*)(&Vt[t * 16 + lm][lh * 16 + 8]);
            o[t] = wmma_bf16(pf, vf, o[t]);
        }
        __syncthreads();
    }

    // ---- normalize + store bf16 in [B*S, 2048] layout ----
#pragma unroll
    for (int t = 0; t < 8; ++t) {
#pragma unroll
        for (int i = 0; i < 8; ++i) {
            const int srow = qbase + i + (lh << 3);
            const int col  = h * DK + t * 16 + lm;
            const float val = o[t][i] / li[i];
            O[((size_t)b * S + srow) * D_MODEL + col] = f2bfu(val);
        }
    }
}

// ---------------------------------------------------------------------------
// launch
// ---------------------------------------------------------------------------
extern "C" void kernel_launch(void* const* d_in, const int* in_sizes, int n_in,
                              void* d_out, int out_size, void* d_ws, size_t ws_size,
                              hipStream_t stream)
{
    const int B = 2, S = 2048, D = D_MODEL;
    const int M = B * S;

    const float* X    = (const float*)d_in[0];
    const float* Wq   = (const float*)d_in[1];
    const float* Wqb  = (const float*)d_in[2];
    const float* Wo   = (const float*)d_in[3];
    const float* Wob  = (const float*)d_in[4];
    const float* rmsw = (const float*)d_in[5];
    float* out = (float*)d_out;

    char* p = (char*)d_ws;
    auto carve = [&](size_t bytes) -> char* {
        char* r = p;
        p += (bytes + 255) & ~(size_t)255;
        return r;
    };
    unsigned short* Xn  = (unsigned short*)carve((size_t)M * D * 2);
    unsigned short* WqB = (unsigned short*)carve((size_t)D * D * 2);
    unsigned short* WoB = (unsigned short*)carve((size_t)D * D * 2);
    float*          Qf  = (float*)carve((size_t)M * D * 4);
    unsigned short* Qr  = (unsigned short*)carve((size_t)M * D * 2);
    unsigned short* Vb  = (unsigned short*)carve((size_t)M * D * 2);
    unsigned short* Oa  = (unsigned short*)carve((size_t)M * D * 2);
    (void)ws_size; (void)in_sizes; (void)n_in; (void)out_size;

    cast_f32_bf16<<<(D * D + 255) / 256, 256, 0, stream>>>(Wq, WqB, D * D);
    cast_f32_bf16<<<(D * D + 255) / 256, 256, 0, stream>>>(Wo, WoB, D * D);
    rmsnorm_to_bf16<<<M, 256, 0, stream>>>(X, rmsw, Xn, D);
    gemm_bf16_nt<<<dim3(D / 256, M / 64), 256, 0, stream>>>(
        Xn, WqB, Wqb, nullptr, Qf, M, D, D, 0);
    rope_pack<<<(int)(((size_t)M * D) / 256), 256, 0, stream>>>(Qf, Qr, Vb, S);
    flash_attn<<<dim3(S / 16, B * NHEAD), 32, 0, stream>>>(Qr, Vb, Oa, S);
    gemm_bf16_nt<<<dim3(D / 256, M / 64), 256, 0, stream>>>(
        Oa, WoB, Wob, X, out, M, D, D, 1);
}
